// HighOrderDotProductAttention_58256936403215
// MI455X (gfx1250) — matrix operand order repaired and checked by reading, not compiled, not run on hardware
//
#include <hip/hip_runtime.h>

// HighOrderDotProductAttention for MI455X (gfx1250, wave32, WMMA).
// Per (b,i): S = (K ⊙ q_i) Lᵀ  -> masked row softmax over k -> T = P V ->
// out_i[d] = Σ_j V[j,d] T[j,d].  All GEMM work on v_wmma_f32_16x16x32_f16.

typedef __attribute__((ext_vector_type(16))) _Float16 v16h;
typedef __attribute__((ext_vector_type(8)))  _Float16 h8;
typedef __attribute__((ext_vector_type(8)))  float    v8f;

#define BHN 32
#define NSEQ 128
#define DHEAD 64
#define MASKV (-1.0e6f)

// A-operand fragment (16x32 f16): lane half 'hf' holds K = hf*8 + {0..7, 16..23}.
// p points at the first chunk; second chunk is p+16. Both 16B aligned.
__device__ __forceinline__ v16h fragA(const _Float16* p) {
  h8 c0 = *(const h8*)(p);
  h8 c1 = *(const h8*)(p + 16);
  return __builtin_shufflevector(c0, c1, 0,1,2,3,4,5,6,7,8,9,10,11,12,13,14,15);
}

// Unused contiguous loader (B operands use fragA's order): K = hf*16 + {0..15},
// i.e. 16 contiguous halfs at p.
__device__ __forceinline__ v16h fragB(const _Float16* p) {
  h8 c0 = *(const h8*)(p);
  h8 c1 = *(const h8*)(p + 8);
  return __builtin_shufflevector(c0, c1, 0,1,2,3,4,5,6,7,8,9,10,11,12,13,14,15);
}

extern "C" __global__ void __launch_bounds__(256)
hoattn_kernel(const float* __restrict__ q,
              const float* __restrict__ k,
              const float* __restrict__ v,
              const float* __restrict__ l,
              const int*   __restrict__ vlen,
              float* __restrict__ out)
{
  extern __shared__ char smem[];
  float*    sK  = (float*)(smem);                      // 128x64 f32   32KB
  float*    sQ  = (float*)(smem + 32768);              //   8x64 f32    2KB
  _Float16* sL  = (_Float16*)(smem + 34816);           // 128x64 f16   16KB
  _Float16* sVt = (_Float16*)(smem + 34816 + 16384);   //  64x128 f16  16KB (V transposed)
  _Float16* sP  = (_Float16*)(smem + 34816 + 32768);   // 8 x 16x128   32KB

  const int tid   = threadIdx.x;
  const int b     = blockIdx.x >> 4;
  const int ibase = (blockIdx.x & 15) * 8;
  const size_t boff = (size_t)b * (NSEQ * DHEAD);

  // ---- cooperative stage: K (f32), L (f16), V -> Vt (f16), Q rows ----
  const float4* Kg = (const float4*)(k + boff);
  const float4* Lg = (const float4*)(l + boff);
  const float4* Vg = (const float4*)(v + boff);
  #pragma unroll
  for (int it = 0; it < 8; ++it) {
    const int idx = tid + it * 256;          // 2048 float4 per tensor
    ((float4*)sK)[idx] = Kg[idx];
    const float4 lv = Lg[idx];
    const int e = idx * 4;
    sL[e+0] = (_Float16)lv.x; sL[e+1] = (_Float16)lv.y;
    sL[e+2] = (_Float16)lv.z; sL[e+3] = (_Float16)lv.w;
    const float4 vv = Vg[idx];
    const int kr = e >> 6, dc = e & 63;      // V[kr][dc..dc+3] -> Vt[dc][kr]
    sVt[(dc+0)*NSEQ + kr] = (_Float16)vv.x;
    sVt[(dc+1)*NSEQ + kr] = (_Float16)vv.y;
    sVt[(dc+2)*NSEQ + kr] = (_Float16)vv.z;
    sVt[(dc+3)*NSEQ + kr] = (_Float16)vv.w;
  }
  #pragma unroll
  for (int it = 0; it < 2; ++it) {
    const int idx = tid + it * 256;
    sQ[idx] = q[boff + (size_t)ibase * DHEAD + idx];
  }
  const int VL = vlen[b];
  __syncthreads();

  const int wv  = tid >> 5;
  const int ln  = tid & 15;        // lane % 16
  const int hf  = (tid >> 4) & 1;  // lane half
  const int bbA = hf * 8;          // A-operand K base
  const int bbB = hf * 16;         // contiguous K base (unused)
  const int i   = ibase + wv;
  _Float16* Pw = sP + wv * (16 * NSEQ);

  // q_i fragments in A layout (kept f32 for exact scaling of K)
  float qf[2][16];
  #pragma unroll
  for (int s = 0; s < 2; ++s) {
    const float* qp = sQ + wv * DHEAD + s * 32 + bbA;
    #pragma unroll
    for (int t = 0; t < 8; ++t) { qf[s][t] = qp[t]; qf[s][8+t] = qp[16+t]; }
  }

  float outacc[4] = {0.f, 0.f, 0.f, 0.f};

  for (int jt = 0; jt < 8; ++jt) {
    // A fragments: rows of K scaled by q_i, converted to f16
    v16h afr[2];
    const int jrow = jt * 16 + ln;
    #pragma unroll
    for (int s = 0; s < 2; ++s) {
      const float* kp = sK + jrow * DHEAD + s * 32 + bbA;
      #pragma unroll
      for (int t = 0; t < 8; ++t) {
        afr[s][t]   = (_Float16)(kp[t]      * qf[s][t]);
        afr[s][8+t] = (_Float16)(kp[16 + t] * qf[s][8+t]);
      }
    }

    // GEMM1: S row-block (16 x 128), K-dim = 64 (2 WMMA steps per k-tile)
    v8f Sacc[8];
    #pragma unroll
    for (int kt = 0; kt < 8; ++kt) {
      v8f acc = {0.f,0.f,0.f,0.f,0.f,0.f,0.f,0.f};
      const int nrow = kt * 16 + ln;          // column k of S = row of L
      #pragma unroll
      for (int s = 0; s < 2; ++s) {
        v16h bfr = fragA(sL + nrow * DHEAD + s * 32 + bbA);
        acc = __builtin_amdgcn_wmma_f32_16x16x32_f16(false, afr[s], false, bfr,
                                                     (short)0, acc, false, false);
      }
      Sacc[kt] = acc;
    }

    // mask columns k >= valid_len[b]
    #pragma unroll
    for (int kt = 0; kt < 8; ++kt) {
      if (kt * 16 + ln >= VL) {
        #pragma unroll
        for (int r = 0; r < 8; ++r) Sacc[kt][r] = MASKV;
      }
    }

    // row softmax over k: row m = r + 8*hf lives in register r across 16 lanes
    float rinv[8];
    #pragma unroll
    for (int r = 0; r < 8; ++r) {
      float m = Sacc[0][r];
      #pragma unroll
      for (int kt = 1; kt < 8; ++kt) m = fmaxf(m, Sacc[kt][r]);
      #pragma unroll
      for (int sh = 1; sh < 16; sh <<= 1) m = fmaxf(m, __shfl_xor(m, sh, 32));
      float ssum = 0.f;
      #pragma unroll
      for (int kt = 0; kt < 8; ++kt) {
        const float p = __expf(Sacc[kt][r] - m);
        Sacc[kt][r] = p;
        ssum += p;
      }
      #pragma unroll
      for (int sh = 1; sh < 16; sh <<= 1) ssum += __shfl_xor(ssum, sh, 32);
      rinv[r] = 1.0f / ssum;
    }

    // stage P (f16) row-major [m][k] for the A-operand of GEMM2
    #pragma unroll
    for (int kt = 0; kt < 8; ++kt) {
      #pragma unroll
      for (int r = 0; r < 8; ++r) {
        Pw[(r + 8*hf) * NSEQ + kt * 16 + ln] = (_Float16)(Sacc[kt][r] * rinv[r]);
      }
    }

    // GEMM2: T = P (16x128) @ V (128x64), fold into out via V[j,d]*T[j,d]
    #pragma unroll
    for (int nt = 0; nt < 4; ++nt) {
      v8f T = {0.f,0.f,0.f,0.f,0.f,0.f,0.f,0.f};
      #pragma unroll
      for (int s2 = 0; s2 < 4; ++s2) {
        v16h a2 = fragA(Pw + ln * NSEQ + s2 * 32 + bbA);
        v16h b2 = fragA(sVt + (nt * 16 + ln) * NSEQ + s2 * 32 + bbA);
        T = __builtin_amdgcn_wmma_f32_16x16x32_f16(false, a2, false, b2,
                                                   (short)0, T, false, false);
      }
      const int dcol = nt * 16 + ln;
      #pragma unroll
      for (int r = 0; r < 8; ++r) {
        const int j = jt * 16 + r + 8*hf;
        outacc[nt] += (float)sVt[dcol * NSEQ + j] * T[r];
      }
    }
  }

  // combine the two half-wave partial sums (rows 0-7 vs 8-15) and store
  #pragma unroll
  for (int nt = 0; nt < 4; ++nt) {
    const float tot = outacc[nt] + __shfl_xor(outacc[nt], 16, 32);
    if (hf == 0) out[boff + (size_t)i * DHEAD + nt * 16 + ln] = tot;
  }
}

extern "C" void kernel_launch(void* const* d_in, const int* in_sizes, int n_in,
                              void* d_out, int out_size, void* d_ws, size_t ws_size,
                              hipStream_t stream) {
  const float* q = (const float*)d_in[0];
  const float* k = (const float*)d_in[1];
  const float* v = (const float*)d_in[2];
  const float* l = (const float*)d_in[3];
  const int*  vl = (const int*)d_in[4];
  float* out = (float*)d_out;
  (void)in_sizes; (void)n_in; (void)out_size; (void)d_ws; (void)ws_size;

  const size_t shmem = 32768 + 2048 + 16384 + 16384 + 32768; // 100352 B
  hipLaunchKernelGGL(hoattn_kernel, dim3(BHN * 16), dim3(256), shmem, stream,
                     q, k, v, l, vl, out);
}
